// MultiHeadAttention_6794638262367
// MI455X (gfx1250) — matrix operand order repaired and checked by reading, not compiled, not run on hardware
//
#include <hip/hip_runtime.h>
#include <hip/hip_bf16.h>

#define DI __device__ __forceinline__

typedef __bf16 v16bf __attribute__((ext_vector_type(16)));
typedef __bf16 v8bf  __attribute__((ext_vector_type(8)));
typedef float  v8f   __attribute__((ext_vector_type(8)));

constexpr int Bq = 4;     // batch
constexpr int Sq = 2048;  // sequence
constexpr int Eq = 1024;  // embed
constexpr int Hq = 16;    // heads
constexpr int Dq = 64;    // head size

union Frag { v16bf v; v8bf h[2]; unsigned short u[16]; };

DI unsigned short f2bf(float f) {
  unsigned int u = __float_as_uint(f);
  u += 0x7FFFu + ((u >> 16) & 1u);   // round-to-nearest-even
  return (unsigned short)(u >> 16);
}

// ---------------------------------------------------------------- converts
// 4 elements per thread: float4 load, 4x bf16 packed store.
__global__ void k_convert(const float* __restrict__ src,
                          unsigned short* __restrict__ dst, int n4, float scale) {
  int i = blockIdx.x * blockDim.x + threadIdx.x;
  if (i < n4) {
    float4 f = ((const float4*)src)[i];
    union { unsigned int w[2]; unsigned short u[4]; } o;
    o.u[0] = f2bf(f.x * scale); o.u[1] = f2bf(f.y * scale);
    o.u[2] = f2bf(f.z * scale); o.u[3] = f2bf(f.w * scale);
    ((uint2*)dst)[i] = make_uint2(o.w[0], o.w[1]);
  }
}

// WoT[k*E + e] = Wo[e*E + k]  (so out-proj is row-major A·B)
__global__ void k_transpose_wo(const float* __restrict__ wo,
                               unsigned short* __restrict__ wot) {
  int i = blockIdx.x * blockDim.x + threadIdx.x;
  if (i < Eq * Eq) {
    int k = i >> 10, e = i & 1023;
    wot[i] = f2bf(wo[e * Eq + k]);
  }
}

// ---------------------------------------------------------------- lane reductions
// C/D rows live across 16-lane half-waves: reduce within each half.
DI float half_max(float x) {
#pragma unroll
  for (int m = 8; m >= 1; m >>= 1) x = fmaxf(x, __shfl_xor(x, m, 32));
  return x;
}
DI float half_sum(float x) {
#pragma unroll
  for (int m = 8; m >= 1; m >>= 1) x += __shfl_xor(x, m, 32);
  return x;
}

// ---------------------------------------------------------------- QKV projection GEMM
// C[z][M,64] = X[b][M,K=1024] * W[h][K,64]  (bf16 in/out, f32 acc)
// 256 threads = 8 waves; block tile M=256, N=64, K-step 32, double-buffered LDS.
__global__ __launch_bounds__(256) void k_gemm_qkv(
    const unsigned short* __restrict__ Xbf,   // [B, S, E]
    const unsigned short* __restrict__ Wbf,   // [H, E, D]
    unsigned short* __restrict__ Out)         // [B*H, S, D]
{
  __shared__ alignas(32) unsigned short Atile[2][256 * 32];  // [m][k]
  __shared__ alignas(32) unsigned short BtT[2][64 * 32];     // [n][k]
  const int tid = threadIdx.x;
  const int wave = tid >> 5, lane = tid & 31;
  const int row = lane & 15, hi = lane >> 4;
  const int m0 = blockIdx.x * 256;
  const int z = blockIdx.y;
  const int b = z / Hq, h = z % Hq;
  const unsigned short* A  = Xbf + (size_t)b * Sq * Eq;
  const unsigned short* Bm = Wbf + (size_t)h * Eq * Dq;
  unsigned short* C = Out + (size_t)z * Sq * Dq;

  const int brow = tid >> 3, bcol = (tid & 7) << 3;

  auto stage = [&](int buf, int kk) {
    { // A: one full 32-col row per thread (64B contiguous)
      const uint4* g = (const uint4*)(A + (size_t)(m0 + tid) * Eq + kk);
      uint4* s = (uint4*)(Atile[buf] + tid * 32);
      s[0] = g[0]; s[1] = g[1]; s[2] = g[2]; s[3] = g[3];
    }
    { // B: 32x64 -> transposed [n][k]
      union { uint4 q; unsigned short u[8]; } t;
      t.q = *(const uint4*)(Bm + (size_t)(kk + brow) * Dq + bcol);
#pragma unroll
      for (int i = 0; i < 8; ++i) BtT[buf][(bcol + i) * 32 + brow] = t.u[i];
    }
  };

  v8f acc[2][4];
#pragma unroll
  for (int mt = 0; mt < 2; ++mt)
#pragma unroll
    for (int nt = 0; nt < 4; ++nt)
#pragma unroll
      for (int r = 0; r < 8; ++r) acc[mt][nt][r] = 0.f;

  stage(0, 0);
  __syncthreads();

  for (int kk = 0; kk < Eq; kk += 32) {
    const int cur = (kk >> 5) & 1;
    if (kk + 32 < Eq) stage(cur ^ 1, kk + 32);   // overlap next tile with compute

    // batch all fragment loads, then issue WMMAs back-to-back
    Frag af[2], bf[4];
#pragma unroll
    for (int mt = 0; mt < 2; ++mt) {
      const unsigned short* ap =
          Atile[cur] + (wave * 32 + mt * 16 + row) * 32 + (hi ? 16 : 0);
      af[mt].h[0] = *(const v8bf*)ap;
      af[mt].h[1] = *(const v8bf*)(ap + 8);
    }
#pragma unroll
    for (int nt = 0; nt < 4; ++nt)
      bf[nt].v = *(const v16bf*)(BtT[cur] + (nt * 16 + row) * 32 + (hi ? 16 : 0));
#pragma unroll
    for (int mt = 0; mt < 2; ++mt)
#pragma unroll
      for (int nt = 0; nt < 4; ++nt)
        acc[mt][nt] = __builtin_amdgcn_wmma_f32_16x16x32_bf16(
            false, af[mt].v, false, bf[nt].v, (short)0, acc[mt][nt], false, false);
    __syncthreads();
  }

#pragma unroll
  for (int mt = 0; mt < 2; ++mt)
#pragma unroll
    for (int nt = 0; nt < 4; ++nt)
#pragma unroll
      for (int r = 0; r < 8; ++r) {
        int m = m0 + wave * 32 + mt * 16 + (hi ? 8 + r : r);
        C[(size_t)m * Dq + nt * 16 + row] = f2bf(acc[mt][nt][r]);
      }
}

// ---------------------------------------------------------------- flash attention
// 128 threads = 4 waves; 64 query rows per block; 32-key blocks, double-buffered.
__global__ __launch_bounds__(128) void k_attn(
    const unsigned short* __restrict__ Qbf,   // [B*H, S, D] (pre-scaled by 1/8)
    const unsigned short* __restrict__ Kbf,
    const unsigned short* __restrict__ Vbf,
    unsigned short* __restrict__ AO)          // [B, S, E]
{
  __shared__ alignas(32) unsigned short Klds[2][32 * 64];    // [key][d]
  __shared__ alignas(32) unsigned short Vt[2][64 * 32];      // [d][key]
  __shared__ alignas(32) unsigned short Plds[4 * 16 * 32];   // per-wave P scratch
  const int tid = threadIdx.x;
  const int wave = tid >> 5, lane = tid & 31;
  const int row = lane & 15, hi = lane >> 4;
  const int q0 = blockIdx.x * 64;
  const int h = blockIdx.y, b = blockIdx.z;
  const int z = b * Hq + h;
  const unsigned short* Qp = Qbf + (size_t)z * Sq * Dq;
  const unsigned short* Kp = Kbf + (size_t)z * Sq * Dq;
  const unsigned short* Vp = Vbf + (size_t)z * Sq * Dq;

  const int sr = tid >> 2, sc = (tid & 3) << 4;  // staging split (32 rows x 64 cols)

  auto stageKV = [&](int buf, int jbase) {
    { // K row-major
      const uint4* g = (const uint4*)(Kp + (size_t)(jbase + sr) * Dq + sc);
      uint4* s = (uint4*)(Klds[buf] + sr * 64 + sc);
      s[0] = g[0]; s[1] = g[1];
    }
    { // V transposed -> Vt[d][key]
      union { uint4 q[2]; unsigned short u[16]; } t;
      const uint4* g = (const uint4*)(Vp + (size_t)(jbase + sr) * Dq + sc);
      t.q[0] = g[0]; t.q[1] = g[1];
#pragma unroll
      for (int i = 0; i < 16; ++i) Vt[buf][(sc + i) * 32 + sr] = t.u[i];
    }
  };

  Frag qf[2];  // Q rows for this wave, d-chunks 0..31 / 32..63
  {
    const int qrow = q0 + wave * 16 + row;
#pragma unroll
    for (int ch = 0; ch < 2; ++ch) {
      const unsigned short* p = Qp + (size_t)qrow * Dq + ch * 32 + (hi ? 16 : 0);
      qf[ch].h[0] = *(const v8bf*)p;
      qf[ch].h[1] = *(const v8bf*)(p + 8);
    }
  }

  v8f oacc[4];
#pragma unroll
  for (int nt = 0; nt < 4; ++nt)
#pragma unroll
    for (int r = 0; r < 8; ++r) oacc[nt][r] = 0.f;
  float mrow[8], lrow[8];
#pragma unroll
  for (int r = 0; r < 8; ++r) { mrow[r] = -1e30f; lrow[r] = 0.f; }

  const int rbase = q0 + wave * 16 + (hi ? 8 : 0);
  const int nkb = q0 / 32 + 2;              // causal: keys <= q0+63

  stageKV(0, 0);
  __syncthreads();

  for (int jb = 0; jb < nkb; ++jb) {
    const int jbase = jb * 32;
    const int cur = jb & 1;
    if (jb + 1 < nkb) stageKV(cur ^ 1, jbase + 32);

    // S = Q * K^T : 16x32 scores, contraction over Dh=64 (2 chunks)
    Frag kb[2][2];
#pragma unroll
    for (int nt = 0; nt < 2; ++nt)
#pragma unroll
      for (int ch = 0; ch < 2; ++ch)
        kb[nt][ch].v = *(const v16bf*)(Klds[cur] + (nt * 16 + row) * 64 +
                                       ch * 32 + (hi ? 16 : 0));
    v8f sacc[2];
#pragma unroll
    for (int nt = 0; nt < 2; ++nt)
#pragma unroll
      for (int r = 0; r < 8; ++r) sacc[nt][r] = 0.f;
#pragma unroll
    for (int nt = 0; nt < 2; ++nt)
#pragma unroll
      for (int ch = 0; ch < 2; ++ch)
        sacc[nt] = __builtin_amdgcn_wmma_f32_16x16x32_bf16(
            false, qf[ch].v, false, kb[nt][ch].v, (short)0, sacc[nt], false, false);

    // causal mask + online softmax (rows spread across half-wave lanes)
    float pv0[8], pv1[8], corr[8];
#pragma unroll
    for (int r = 0; r < 8; ++r) {
      const int rg = rbase + r;
      float s0 = (jbase + row      > rg) ? -1e30f : sacc[0][r];
      float s1 = (jbase + 16 + row > rg) ? -1e30f : sacc[1][r];
      float mx = half_max(fmaxf(s0, s1));
      float mnew = fmaxf(mrow[r], mx);
      corr[r] = __expf(mrow[r] - mnew);
      float p0 = __expf(s0 - mnew);
      float p1 = __expf(s1 - mnew);
      lrow[r] = lrow[r] * corr[r] + half_sum(p0 + p1);
      mrow[r] = mnew;
      pv0[r] = p0; pv1[r] = p1;
    }

    // C-layout -> A-layout via per-wave LDS scratch (wave-local, no barrier)
    unsigned short* Pw = Plds + wave * (16 * 32);
#pragma unroll
    for (int r = 0; r < 8; ++r) {
      const int rl = (hi ? 8 + r : r);
      Pw[rl * 32 + row]      = f2bf(pv0[r]);
      Pw[rl * 32 + 16 + row] = f2bf(pv1[r]);
    }
#pragma unroll
    for (int nt = 0; nt < 4; ++nt)
#pragma unroll
      for (int r = 0; r < 8; ++r) oacc[nt][r] *= corr[r];

    Frag pa, vb[4];
    {
      const unsigned short* p = Pw + row * 32 + (hi ? 16 : 0);
      pa.h[0] = *(const v8bf*)p;
      pa.h[1] = *(const v8bf*)(p + 8);
    }
#pragma unroll
    for (int nt = 0; nt < 4; ++nt)
      vb[nt].v = *(const v16bf*)(Vt[cur] + (nt * 16 + row) * 32 + (hi ? 16 : 0));
    // O += P * V : contraction over 32 keys, 4 output n-tiles of Dh
#pragma unroll
    for (int nt = 0; nt < 4; ++nt)
      oacc[nt] = __builtin_amdgcn_wmma_f32_16x16x32_bf16(
          false, pa.v, false, vb[nt].v, (short)0, oacc[nt], false, false);

    __syncthreads();  // all waves done with buf `cur` before it is restaged
  }

  float invl[8];
#pragma unroll
  for (int r = 0; r < 8; ++r) invl[r] = 1.f / lrow[r];
#pragma unroll
  for (int nt = 0; nt < 4; ++nt)
#pragma unroll
    for (int r = 0; r < 8; ++r) {
      const int srow = q0 + wave * 16 + (hi ? 8 + r : r);
      AO[((size_t)b * Sq + srow) * Eq + h * Dq + nt * 16 + row] =
          f2bf(oacc[nt][r] * invl[r]);
    }
}

// ---------------------------------------------------------------- output projection
// Out[M,1024] = Attn[M,1024] * WoT[1024,1024] + bo   (f32 out)
__global__ __launch_bounds__(256) void k_gemm_out(
    const unsigned short* __restrict__ Abf,  // [B*S, E]
    const unsigned short* __restrict__ WoT,  // [E, E]  (K rows x N cols)
    const float* __restrict__ bias,          // [E]
    float* __restrict__ Out)                 // [B*S, E]
{
  __shared__ alignas(32) unsigned short Atile[2][256 * 32];
  __shared__ alignas(32) unsigned short BtT[2][64 * 32];
  const int tid = threadIdx.x;
  const int wave = tid >> 5, lane = tid & 31;
  const int row = lane & 15, hi = lane >> 4;
  const int m0 = blockIdx.x * 256;
  const int n0 = blockIdx.y * 64;

  const int brow = tid >> 3, bcol = (tid & 7) << 3;

  auto stage = [&](int buf, int kk) {
    {
      const uint4* g = (const uint4*)(Abf + (size_t)(m0 + tid) * Eq + kk);
      uint4* s = (uint4*)(Atile[buf] + tid * 32);
      s[0] = g[0]; s[1] = g[1]; s[2] = g[2]; s[3] = g[3];
    }
    {
      union { uint4 q; unsigned short u[8]; } t;
      t.q = *(const uint4*)(WoT + (size_t)(kk + brow) * Eq + n0 + bcol);
#pragma unroll
      for (int i = 0; i < 8; ++i) BtT[buf][(bcol + i) * 32 + brow] = t.u[i];
    }
  };

  v8f acc[2][4];
#pragma unroll
  for (int mt = 0; mt < 2; ++mt)
#pragma unroll
    for (int nt = 0; nt < 4; ++nt)
#pragma unroll
      for (int r = 0; r < 8; ++r) acc[mt][nt][r] = 0.f;

  stage(0, 0);
  __syncthreads();

  for (int kk = 0; kk < Eq; kk += 32) {
    const int cur = (kk >> 5) & 1;
    if (kk + 32 < Eq) stage(cur ^ 1, kk + 32);

    Frag af[2], bf[4];
#pragma unroll
    for (int mt = 0; mt < 2; ++mt) {
      const unsigned short* ap =
          Atile[cur] + (wave * 32 + mt * 16 + row) * 32 + (hi ? 16 : 0);
      af[mt].h[0] = *(const v8bf*)ap;
      af[mt].h[1] = *(const v8bf*)(ap + 8);
    }
#pragma unroll
    for (int nt = 0; nt < 4; ++nt)
      bf[nt].v = *(const v16bf*)(BtT[cur] + (nt * 16 + row) * 32 + (hi ? 16 : 0));
#pragma unroll
    for (int mt = 0; mt < 2; ++mt)
#pragma unroll
      for (int nt = 0; nt < 4; ++nt)
        acc[mt][nt] = __builtin_amdgcn_wmma_f32_16x16x32_bf16(
            false, af[mt].v, false, bf[nt].v, (short)0, acc[mt][nt], false, false);
    __syncthreads();
  }

#pragma unroll
  for (int mt = 0; mt < 2; ++mt)
#pragma unroll
    for (int nt = 0; nt < 4; ++nt)
#pragma unroll
      for (int r = 0; r < 8; ++r) {
        int m = m0 + wave * 32 + mt * 16 + (hi ? 8 + r : r);
        int n = n0 + nt * 16 + row;
        Out[(size_t)m * Eq + n] = acc[mt][nt][r] + bias[n];
      }
}

// ---------------------------------------------------------------- launch
extern "C" void kernel_launch(void* const* d_in, const int* in_sizes, int n_in,
                              void* d_out, int out_size, void* d_ws, size_t ws_size,
                              hipStream_t stream) {
  (void)in_sizes; (void)n_in; (void)out_size; (void)ws_size;
  const float* X  = (const float*)d_in[0];
  const float* Wq = (const float*)d_in[1];
  const float* Wk = (const float*)d_in[2];
  const float* Wv = (const float*)d_in[3];
  const float* Wo = (const float*)d_in[4];
  const float* bo = (const float*)d_in[5];
  float* out = (float*)d_out;

  // workspace carve-up (bf16 buffers, 256B aligned); ~72 MB total
  char* ws = (char*)d_ws;
  size_t off = 0;
  auto take = [&](size_t elems) {
    unsigned short* p = (unsigned short*)(ws + off);
    off += ((elems * 2) + 255) & ~(size_t)255;
    return p;
  };
  unsigned short* Xbf  = take((size_t)Bq * Sq * Eq);
  unsigned short* Wqbf = take((size_t)Hq * Eq * Dq);
  unsigned short* Wkbf = take((size_t)Hq * Eq * Dq);
  unsigned short* Wvbf = take((size_t)Hq * Eq * Dq);
  unsigned short* WoT  = take((size_t)Eq * Eq);
  unsigned short* Qbf  = take((size_t)Bq * Hq * Sq * Dq);
  unsigned short* Kbf  = take((size_t)Bq * Hq * Sq * Dq);
  unsigned short* Vbf  = take((size_t)Bq * Hq * Sq * Dq);
  unsigned short* Attn = Xbf;  // X dead after QKV projections -> reuse

  const int nX4 = (Bq * Sq * Eq) / 4;
  const int nW4 = (Hq * Eq * Dq) / 4;
  k_convert<<<(nX4 + 255) / 256, 256, 0, stream>>>(X, Xbf, nX4, 1.0f);
  k_convert<<<(nW4 + 255) / 256, 256, 0, stream>>>(Wq, Wqbf, nW4, 0.125f); // fold 1/sqrt(Dh)
  k_convert<<<(nW4 + 255) / 256, 256, 0, stream>>>(Wk, Wkbf, nW4, 1.0f);
  k_convert<<<(nW4 + 255) / 256, 256, 0, stream>>>(Wv, Wvbf, nW4, 1.0f);
  k_transpose_wo<<<(Eq * Eq + 255) / 256, 256, 0, stream>>>(Wo, WoT);

  dim3 gq(Sq / 256, Bq * Hq);
  k_gemm_qkv<<<gq, 256, 0, stream>>>(Xbf, Wqbf, Qbf);
  k_gemm_qkv<<<gq, 256, 0, stream>>>(Xbf, Wkbf, Kbf);
  k_gemm_qkv<<<gq, 256, 0, stream>>>(Xbf, Wvbf, Vbf);

  dim3 ga(Sq / 64, Hq, Bq);
  k_attn<<<ga, 128, 0, stream>>>(Qbf, Kbf, Vbf, Attn);

  dim3 go((Bq * Sq) / 256, Eq / 64);
  k_gemm_out<<<go, 256, 0, stream>>>(Attn, WoT, bo, out);
}
